// DNCMemorySystem_70385924047062
// MI455X (gfx1250) — compile-verified, matrix operand order checked by reading, not run on hardware
//
#include <hip/hip_runtime.h>
#include <hip/hip_bf16.h>
#include <math.h>

// ---------------- problem dims ----------------
#define Bq   64
#define Nq   1024
#define Wq   64
#define Rq   4
#define Hq   256
#define INq  320          // W + R*W
#define IFq  471          // interface vector length
#define IFP  480          // padded interface length (multiple of 16)

// ---------------- WMMA types ------------------
typedef __attribute__((ext_vector_type(16))) __bf16 v16bf;
typedef __attribute__((ext_vector_type(8)))  float  v8f;

union ABf { v16bf v; unsigned u[8]; };

__device__ inline unsigned pack2bf(float x, float y) {
    unsigned a = __float_as_uint(x);
    unsigned b = __float_as_uint(y);
    return (a >> 16) | (b & 0xFFFF0000u);
}

__device__ inline float sigmoidf_(float x) { return 1.f / (1.f + __expf(-x)); }
__device__ inline float softplusf_(float x) { return (x > 20.f) ? x : log1pf(__expf(x)); }

// ------------- block reductions (wave32) -------------
__device__ inline float breduce_sum(float v, float* red) {
    for (int o = 16; o > 0; o >>= 1) v += __shfl_xor(v, o);
    int wid = threadIdx.x >> 5, lid = threadIdx.x & 31;
    int nw = blockDim.x >> 5;
    if (lid == 0) red[wid] = v;
    __syncthreads();
    if (wid == 0) {
        float r = (lid < nw) ? red[lid] : 0.f;
        for (int o = 16; o > 0; o >>= 1) r += __shfl_xor(r, o);
        if (lid == 0) red[0] = r;
    }
    __syncthreads();
    float out = red[0];
    __syncthreads();
    return out;
}

__device__ inline float breduce_max(float v, float* red) {
    for (int o = 16; o > 0; o >>= 1) v = fmaxf(v, __shfl_xor(v, o));
    int wid = threadIdx.x >> 5, lid = threadIdx.x & 31;
    int nw = blockDim.x >> 5;
    if (lid == 0) red[wid] = v;
    __syncthreads();
    if (wid == 0) {
        float r = (lid < nw) ? red[lid] : -3.4e38f;
        for (int o = 16; o > 0; o >>= 1) r = fmaxf(r, __shfl_xor(r, o));
        if (lid == 0) red[0] = r;
    }
    __syncthreads();
    float out = red[0];
    __syncthreads();
    return out;
}

// =====================================================
// WMMA GEMM (tile-exact, branch-free):
//   Y[M,Nout] = X[M,K] @ Wt[Nout,K]^T + b0 + b1
// Preconditions (guaranteed by launcher): M %16==0, Nout %16==0, K %32==0.
// One wave per 16x16 tile. Per lane, each K-step's fragment is two
// contiguous 8-float runs -> four global_load_b128 per matrix.
// =====================================================
__global__ void wmma_gemm(const float* __restrict__ X,
                          const float* __restrict__ Wt,
                          const float* __restrict__ bias0,
                          const float* __restrict__ bias1,
                          float* __restrict__ Y,
                          int M, int Nout, int K) {
    const int lane = threadIdx.x;
    const int half = lane >> 4;        // ISA 16-bit A/B K-half select
    const int lm   = lane & 15;
    const int row  = blockIdx.y * 16 + lm;   // A row (M)
    const int col  = blockIdx.x * 16 + lm;   // B col (N)

    const float* xr = X  + (size_t)row * K + half * 8;
    const float* wr = Wt + (size_t)col * K + half * 8;

    float bv = 0.f;
    if (bias0) bv += bias0[col];
    if (bias1) bv += bias1[col];

    v8f acc = {};
    for (int k0 = 0; k0 < K; k0 += 32) {
        float4 x0 = *(const float4*)(xr + k0);
        float4 x1 = *(const float4*)(xr + k0 + 4);
        float4 x2 = *(const float4*)(xr + k0 + 16);
        float4 x3 = *(const float4*)(xr + k0 + 20);
        float4 w0 = *(const float4*)(wr + k0);
        float4 w1 = *(const float4*)(wr + k0 + 4);
        float4 w2 = *(const float4*)(wr + k0 + 16);
        float4 w3 = *(const float4*)(wr + k0 + 20);
        ABf a, b;
        a.u[0] = pack2bf(x0.x, x0.y); a.u[1] = pack2bf(x0.z, x0.w);
        a.u[2] = pack2bf(x1.x, x1.y); a.u[3] = pack2bf(x1.z, x1.w);
        a.u[4] = pack2bf(x2.x, x2.y); a.u[5] = pack2bf(x2.z, x2.w);
        a.u[6] = pack2bf(x3.x, x3.y); a.u[7] = pack2bf(x3.z, x3.w);
        b.u[0] = pack2bf(w0.x, w0.y); b.u[1] = pack2bf(w0.z, w0.w);
        b.u[2] = pack2bf(w1.x, w1.y); b.u[3] = pack2bf(w1.z, w1.w);
        b.u[4] = pack2bf(w2.x, w2.y); b.u[5] = pack2bf(w2.z, w2.w);
        b.u[6] = pack2bf(w3.x, w3.y); b.u[7] = pack2bf(w3.z, w3.w);
        acc = __builtin_amdgcn_wmma_f32_16x16x32_bf16(false, a.v, false, b.v,
                                                      (short)0, acc, false, false);
    }
#pragma unroll
    for (int v = 0; v < 8; ++v) {
        int m = blockIdx.y * 16 + v + half * 8;
        Y[(size_t)m * Nout + col] = acc[v] + bv;
    }
}

// -------- zero-pad Wint (471x256 -> 480x256) and bint (471 -> 480) --------
__global__ void k_padW(const float* __restrict__ Wsrc, const float* __restrict__ bsrc,
                       float* __restrict__ Wdst, float* __restrict__ bdst) {
    int idx = blockIdx.x * 256 + threadIdx.x;   // covers IFP*Hq
    if (idx < IFP * Hq) {
        int r = idx >> 8;                       // /Hq
        Wdst[idx] = (r < IFq) ? Wsrc[idx] : 0.f;
    }
    if (idx < IFP) bdst[idx] = (idx < IFq) ? bsrc[idx] : 0.f;
}

// -------- concat [input_vector | read_vectors] --------
__global__ void k_concat(const float* __restrict__ inp, const float* __restrict__ rvec,
                         float* __restrict__ x) {
    int idx = blockIdx.x * 256 + threadIdx.x;
    if (idx >= Bq * INq) return;
    int b = idx / INq, c = idx % INq;
    x[idx] = (c < Wq) ? inp[b * Wq + c] : rvec[b * Rq * Wq + (c - Wq)];
}

// -------- LSTM gate nonlinearity (no cell state carry) --------
__global__ void k_lstm_act(const float* __restrict__ g, float* __restrict__ h) {
    int idx = blockIdx.x * 256 + threadIdx.x;
    if (idx >= Bq * Hq) return;
    int b = idx >> 8, j = idx & 255;
    const float* gb = g + (size_t)b * 4 * Hq;
    float i_ = sigmoidf_(gb[j]);
    float gg = tanhf(gb[2 * Hq + j]);
    float o_ = sigmoidf_(gb[3 * Hq + j]);
    h[idx] = o_ * tanhf(i_ * gg);
}

// -------- parse interface vector (stride IFP) --------
__global__ void k_interface(const float* __restrict__ iv,
                            float* rk, float* wk, float* wsp, float* ev, float* wv,
                            float* rs, float* rm, float* wg, float* ag, float* fg) {
    int b = blockIdx.x, t = threadIdx.x;  // 512 threads
    const float* p = iv + (size_t)b * IFP;
    if (t < 256)        rk[b * 256 + t] = p[t];
    else if (t < 320)   wk[b * 64 + (t - 256)] = p[t];
    else if (t == 320)  wsp[b] = softplusf_(p[320]);
    else if (t < 385)   ev[b * 64 + (t - 321)] = sigmoidf_(p[t]);
    else if (t < 449)   wv[b * 64 + (t - 385)] = p[t];
    else if (t < 453)   rs[b * 4 + (t - 449)] = softplusf_(p[t]);
    else if (t < 457) {                       // read-mode softmax rows
        int r = t - 453;
        float a0 = p[453 + 3 * r], a1 = p[454 + 3 * r], a2 = p[455 + 3 * r];
        float mx = fmaxf(a0, fmaxf(a1, a2));
        float e0 = __expf(a0 - mx), e1 = __expf(a1 - mx), e2 = __expf(a2 - mx);
        float s = e0 + e1 + e2;
        rm[b * 12 + 3 * r + 0] = e0 / s;
        rm[b * 12 + 3 * r + 1] = e1 / s;
        rm[b * 12 + 3 * r + 2] = e2 / s;
    }
    else if (t == 465)  wg[b] = sigmoidf_(p[465]);
    else if (t == 466)  ag[b] = sigmoidf_(p[466]);
    else if (t < 471)   fg[b * 4 + (t - 467)] = sigmoidf_(p[t]);
}

// -------- usage update --------
__global__ void k_usage(const float* __restrict__ uin, const float* __restrict__ rw,
                        const float* __restrict__ fg, float* __restrict__ uout) {
    int b = blockIdx.x, n = threadIdx.x;  // 1024
    float prod = 1.f;
#pragma unroll
    for (int r = 0; r < Rq; ++r)
        prod *= (1.f - fg[b * Rq + r] * rw[((size_t)b * Rq + r) * Nq + n]);
    uout[(size_t)b * Nq + n] = uin[(size_t)b * Nq + n] * prod;
}

// -------- content addressing: softmax(scale * cos_sim(key, mem)) --------
__global__ void k_content(const float* __restrict__ keys, const float* __restrict__ scales,
                          const float* __restrict__ mem, float* __restrict__ out, int NK) {
    int bk = blockIdx.x;        // b*NK + k
    int n  = threadIdx.x;       // 1024
    int b  = bk / NK;
    __shared__ float ks[Wq];
    __shared__ float red[32];
    if (n < Wq) ks[n] = keys[bk * Wq + n];
    __syncthreads();
    const float* mrow = mem + ((size_t)b * Nq + n) * Wq;
    float dot = 0.f, nm = 0.f, nk = 0.f;
#pragma unroll 8
    for (int w = 0; w < Wq; ++w) {
        float m = mrow[w], k = ks[w];
        dot += k * m; nm += m * m; nk += k * k;
    }
    float sim = dot / (fmaxf(sqrtf(nk), 1e-8f) * fmaxf(sqrtf(nm), 1e-8f));
    float logit = scales[bk] * sim;
    float mx = breduce_max(logit, red);
    float e  = __expf(logit - mx);
    float s  = breduce_sum(e, red);
    out[(size_t)bk * Nq + n] = e / s;
}

// -------- allocation weights: bitonic sort + prefix-product + scatter --------
__global__ void k_alloc(const float* __restrict__ usage, float* __restrict__ alloc) {
    int b = blockIdx.x, t = threadIdx.x;  // 1024
    __shared__ float key[Nq];
    __shared__ int   idx[Nq];
    __shared__ float scan[Nq];
    key[t] = usage[(size_t)b * Nq + t];
    idx[t] = t;
    __syncthreads();
    for (int ksz = 2; ksz <= Nq; ksz <<= 1) {
        for (int j = ksz >> 1; j > 0; j >>= 1) {
            int p = t ^ j;
            if (p > t) {
                bool up = ((t & ksz) == 0);  // ascending subsequence
                float a = key[t], c = key[p];
                if ((a > c) == up) {
                    key[t] = c; key[p] = a;
                    int ia = idx[t]; idx[t] = idx[p]; idx[p] = ia;
                }
            }
            __syncthreads();
        }
    }
    scan[t] = key[t];
    __syncthreads();
    for (int off = 1; off < Nq; off <<= 1) {
        float v = (t >= off) ? scan[t - off] : 1.f;
        __syncthreads();
        if (t >= off) scan[t] *= v;
        __syncthreads();
    }
    float excl = (t == 0) ? 1.f : scan[t - 1];
    alloc[(size_t)b * Nq + idx[t]] = key[t] * excl;
}

// -------- write weights + precedence update --------
__global__ void k_ww(const float* __restrict__ alloc, const float* __restrict__ cw,
                     const float* __restrict__ wg, const float* __restrict__ ag,
                     const float* __restrict__ prec,
                     float* __restrict__ ww, float* __restrict__ pnew) {
    int b = blockIdx.x, n = threadIdx.x;  // 1024
    __shared__ float red[32];
    float wgv = wg[b], agv = ag[b];
    float w_ = wgv * (agv * alloc[(size_t)b * Nq + n] + (1.f - agv) * cw[(size_t)b * Nq + n]);
    ww[(size_t)b * Nq + n] = w_;
    float s = breduce_sum(w_, red);
    pnew[(size_t)b * Nq + n] = (1.f - s) * prec[(size_t)b * Nq + n] + w_;
}

// -------- memory update --------
__global__ void k_mem(const float* __restrict__ min_, const float* __restrict__ ww,
                      const float* __restrict__ ev, const float* __restrict__ v,
                      float* __restrict__ mout) {
    size_t idx = (size_t)blockIdx.x * 256 + threadIdx.x;
    if (idx >= (size_t)Bq * Nq * Wq) return;
    int w = (int)(idx & 63);
    size_t bn = idx >> 6;
    int n = (int)(bn & 1023);
    int b = (int)(bn >> 10);
    float wwv = ww[(size_t)b * Nq + n];
    mout[idx] = min_[idx] * (1.f - wwv * ev[b * Wq + w]) + wwv * v[b * Wq + w];
}

// =====================================================
// Fused link pass: Lnew write + fwd + bwd in ONE stream of link.
// grid (4, B); block 256; thread t owns row i = tile*256+t of link[b].
// Skewed j walk => bwd LDS accumulation is conflict-free under the
// per-iteration barrier; fwd accumulates in registers (row-exclusive).
// =====================================================
__global__ void k_link(const float* __restrict__ link, const float* __restrict__ ww,
                       const float* __restrict__ prec, const float* __restrict__ rw,
                       float* __restrict__ Lnew, float* __restrict__ fwd,
                       float* __restrict__ bwd) {
    int b  = blockIdx.y;
    int it = blockIdx.x;     // row tile (256 rows)
    int t  = threadIdx.x;    // 256
    __shared__ float ww_s[Nq];
    __shared__ float pr_s[Nq];
    __shared__ float rwj[Rq][Nq];
    __shared__ float bacc[Rq][Nq];
    for (int j = t; j < Nq; j += 256) {
        ww_s[j] = ww[(size_t)b * Nq + j];
        pr_s[j] = prec[(size_t)b * Nq + j];
#pragma unroll
        for (int r = 0; r < Rq; ++r) {
            rwj[r][j]  = rw[((size_t)b * Rq + r) * Nq + j];
            bacc[r][j] = 0.f;
        }
    }
    __syncthreads();

    int i = it * 256 + t;
    float wwi  = ww_s[i];
    float rwi0 = rwj[0][i], rwi1 = rwj[1][i], rwi2 = rwj[2][i], rwi3 = rwj[3][i];
    const float4* lrow = (const float4*)(link + ((size_t)b * Nq + i) * Nq);
    float4*       orow = (float4*)(Lnew + ((size_t)b * Nq + i) * Nq);
    float f0 = 0.f, f1 = 0.f, f2 = 0.f, f3 = 0.f;

    for (int c = 0; c < 256; ++c) {
        int j4 = (t + c) & 255;                  // skewed: distinct j4 per thread
        float4 l4 = lrow[j4];
        __builtin_prefetch(&lrow[(t + c + 16) & 255], 0, 1);
        float o0, o1, o2, o3;
#pragma unroll
        for (int q = 0; q < 4; ++q) {
            int j = j4 * 4 + q;
            float lv = (q == 0) ? l4.x : (q == 1) ? l4.y : (q == 2) ? l4.z : l4.w;
            float val = (1.f - wwi - ww_s[j]) * lv + wwi * pr_s[j];
            if (i == j) val = 0.f;
            if (q == 0) o0 = val; else if (q == 1) o1 = val; else if (q == 2) o2 = val; else o3 = val;
            f0 += val * rwj[0][j]; f1 += val * rwj[1][j];
            f2 += val * rwj[2][j]; f3 += val * rwj[3][j];
            bacc[0][j] += val * rwi0; bacc[1][j] += val * rwi1;
            bacc[2][j] += val * rwi2; bacc[3][j] += val * rwi3;
        }
        orow[j4] = make_float4(o0, o1, o2, o3);
        __syncthreads();                          // keeps skewed j sets disjoint
    }

    fwd[((size_t)b * Rq + 0) * Nq + i] = f0;
    fwd[((size_t)b * Rq + 1) * Nq + i] = f1;
    fwd[((size_t)b * Rq + 2) * Nq + i] = f2;
    fwd[((size_t)b * Rq + 3) * Nq + i] = f3;
    __syncthreads();
    for (int j = t; j < Nq; j += 256)
#pragma unroll
        for (int r = 0; r < Rq; ++r)
            atomicAdd(&bwd[((size_t)b * Rq + r) * Nq + j], bacc[r][j]);
}

// -------- read weights mix --------
__global__ void k_wr(const float* __restrict__ rm, const float* __restrict__ bwd,
                     const float* __restrict__ rcw, const float* __restrict__ fwd,
                     float* __restrict__ wr) {
    int idx = blockIdx.x * 256 + threadIdx.x;    // B*R*N
    if (idx >= Bq * Rq * Nq) return;
    int br = idx >> 10;
    const float* m = rm + br * 3;
    wr[idx] = m[0] * bwd[idx] + m[1] * rcw[idx] + m[2] * fwd[idx];
}

// -------- read vectors rv = wr @ mem --------
__global__ void k_rv(const float* __restrict__ wr, const float* __restrict__ mem,
                     float* __restrict__ rv) {
    int b = blockIdx.x, t = threadIdx.x;  // 256: (r = t>>6, w = t&63)
    __shared__ float wrs[Rq * Nq];
    for (int j = t; j < Rq * Nq; j += 256) wrs[j] = wr[(size_t)b * Rq * Nq + j];
    __syncthreads();
    int r = t >> 6, w = t & 63;
    const float* mb = mem + (size_t)b * Nq * Wq;
    float acc = 0.f;
    for (int n = 0; n < Nq; ++n) acc += wrs[r * Nq + n] * mb[(size_t)n * Wq + w];
    rv[((size_t)b * Rq + r) * Wq + w] = acc;
}

// -------- concat [h | rv] --------
__global__ void k_hv(const float* __restrict__ h, const float* __restrict__ rv,
                     float* __restrict__ hv) {
    int idx = blockIdx.x * 256 + threadIdx.x;   // B*512
    if (idx >= Bq * (Hq + Rq * Wq)) return;
    int b = idx >> 9, c = idx & 511;
    hv[idx] = (c < Hq) ? h[b * Hq + c] : rv[b * Rq * Wq + (c - Hq)];
}

// =====================================================
extern "C" void kernel_launch(void* const* d_in, const int* in_sizes, int n_in,
                              void* d_out, int out_size, void* d_ws, size_t ws_size,
                              hipStream_t stream) {
    const float* input_vector = (const float*)d_in[0];
    const float* memory_mat   = (const float*)d_in[1];
    const float* read_weights = (const float*)d_in[2];
    const float* usage_vec    = (const float*)d_in[3];
    const float* link         = (const float*)d_in[4];
    const float* precedence   = (const float*)d_in[5];
    const float* read_vecs    = (const float*)d_in[6];
    const float* Wih0 = (const float*)d_in[7];
    const float* bih0 = (const float*)d_in[9];
    const float* bhh0 = (const float*)d_in[10];
    const float* Wih1 = (const float*)d_in[11];
    const float* bih1 = (const float*)d_in[13];
    const float* bhh1 = (const float*)d_in[14];
    const float* Wih2 = (const float*)d_in[15];
    const float* bih2 = (const float*)d_in[17];
    const float* bhh2 = (const float*)d_in[18];
    const float* Wint = (const float*)d_in[19];
    const float* bint = (const float*)d_in[20];
    const float* Wenc = (const float*)d_in[21];
    const float* benc = (const float*)d_in[22];
    const float* Wout = (const float*)d_in[23];
    const float* bout = (const float*)d_in[24];

    float* out = (float*)d_out;
    float* ws  = (float*)d_ws;

    // ---- output offsets (floats), in reference return order ----
    const size_t O_OUT   = 0;
    const size_t O_MEM   = 4096;
    const size_t O_WR    = 4198400;
    const size_t O_WW    = 4460544;
    const size_t O_USAGE = 4526080;
    const size_t O_LNEW  = 4591616;
    const size_t O_PNEW  = 71700480;
    const size_t O_RV    = 71766016;

    // ---- workspace layout (floats; every block 16-float aligned) ----
    size_t o = 0;
    const size_t XCAT = o; o += (size_t)Bq * INq;
    const size_t GBUF = o; o += (size_t)Bq * 4 * Hq;
    const size_t HA   = o; o += (size_t)Bq * Hq;
    const size_t HB   = o; o += (size_t)Bq * Hq;
    const size_t IV   = o; o += (size_t)Bq * IFP;
    const size_t WP   = o; o += (size_t)IFP * Hq;   // padded Wint
    const size_t BP   = o; o += IFP;                // padded bint
    const size_t RK   = o; o += (size_t)Bq * Rq * Wq;
    const size_t WK   = o; o += (size_t)Bq * Wq;
    const size_t WSP  = o; o += Bq;
    const size_t EV   = o; o += (size_t)Bq * Wq;
    const size_t WV   = o; o += (size_t)Bq * Wq;
    const size_t RS   = o; o += (size_t)Bq * Rq;
    const size_t RM   = o; o += (size_t)Bq * Rq * 3;
    const size_t WG   = o; o += Bq;
    const size_t AG   = o; o += Bq;
    const size_t FG   = o; o += (size_t)Bq * Rq;
    const size_t CW   = o; o += (size_t)Bq * Nq;
    const size_t ALC  = o; o += (size_t)Bq * Nq;
    const size_t VE   = o; o += (size_t)Bq * Wq;
    const size_t FWD  = o; o += (size_t)Bq * Rq * Nq;
    const size_t BWD  = o; o += (size_t)Bq * Rq * Nq;
    const size_t RCW  = o; o += (size_t)Bq * Rq * Nq;
    const size_t HV   = o; o += (size_t)Bq * (Hq + Rq * Wq);

    // 0) pad interface weights once per call (tiny)
    k_padW<<<(IFP * Hq + 255) / 256, 256, 0, stream>>>(Wint, bint, ws + WP, ws + BP);

    // 1) controller input
    k_concat<<<(Bq * INq + 255) / 256, 256, 0, stream>>>(input_vector, read_vecs, ws + XCAT);

    // 2) 3-layer LSTM stack (branch-free WMMA bf16 GEMMs + gate kernel)
    dim3 gL(4 * Hq / 16, Bq / 16);  // (64, 4)
    wmma_gemm<<<gL, 32, 0, stream>>>(ws + XCAT, Wih0, bih0, bhh0, ws + GBUF, Bq, 4 * Hq, INq);
    k_lstm_act<<<(Bq * Hq + 255) / 256, 256, 0, stream>>>(ws + GBUF, ws + HA);
    wmma_gemm<<<gL, 32, 0, stream>>>(ws + HA, Wih1, bih1, bhh1, ws + GBUF, Bq, 4 * Hq, Hq);
    k_lstm_act<<<(Bq * Hq + 255) / 256, 256, 0, stream>>>(ws + GBUF, ws + HB);
    wmma_gemm<<<gL, 32, 0, stream>>>(ws + HB, Wih2, bih2, bhh2, ws + GBUF, Bq, 4 * Hq, Hq);
    k_lstm_act<<<(Bq * Hq + 255) / 256, 256, 0, stream>>>(ws + GBUF, ws + HA);  // final h

    // 3) interface vector (padded to 480 cols) + parse
    dim3 gI(IFP / 16, Bq / 16);  // (30, 4)
    wmma_gemm<<<gI, 32, 0, stream>>>(ws + HA, ws + WP, ws + BP, nullptr, ws + IV, Bq, IFP, Hq);
    k_interface<<<Bq, 512, 0, stream>>>(ws + IV, ws + RK, ws + WK, ws + WSP, ws + EV,
                                        ws + WV, ws + RS, ws + RM, ws + WG, ws + AG, ws + FG);

    // 4) usage update -> output
    k_usage<<<Bq, Nq, 0, stream>>>(usage_vec, read_weights, ws + FG, out + O_USAGE);

    // 5) write content addressing (OLD memory)
    k_content<<<Bq, Nq, 0, stream>>>(ws + WK, ws + WSP, memory_mat, ws + CW, 1);

    // 6) allocation weights (sort + scan)
    k_alloc<<<Bq, Nq, 0, stream>>>(out + O_USAGE, ws + ALC);

    // 7) write weights + precedence -> outputs
    k_ww<<<Bq, Nq, 0, stream>>>(ws + ALC, ws + CW, ws + WG, ws + AG, precedence,
                                out + O_WW, out + O_PNEW);

    // 8) encoded write vector
    dim3 gE(Wq / 16, Bq / 16);  // (4, 4)
    wmma_gemm<<<gE, 32, 0, stream>>>(ws + WV, Wenc, benc, nullptr, ws + VE, Bq, Wq, Wq);

    // 9) memory update -> output
    k_mem<<<(Bq * Nq * Wq) / 256, 256, 0, stream>>>(memory_mat, out + O_WW, ws + EV,
                                                    ws + VE, out + O_MEM);

    // 10) read content addressing (NEW memory)
    k_content<<<Bq * Rq, Nq, 0, stream>>>(ws + RK, ws + RS, out + O_MEM, ws + RCW, Rq);

    // 11) fused link pass: Lnew + fwd + bwd in one HBM sweep
    hipMemsetAsync(ws + BWD, 0, (size_t)Bq * Rq * Nq * sizeof(float), stream);
    dim3 gLk(4, Bq);
    k_link<<<gLk, 256, 0, stream>>>(link, out + O_WW, precedence, read_weights,
                                    out + O_LNEW, ws + FWD, ws + BWD);

    // 12) read-weight mix -> output
    k_wr<<<(Bq * Rq * Nq) / 256, 256, 0, stream>>>(ws + RM, ws + BWD, ws + RCW, ws + FWD,
                                                   out + O_WR);

    // 13) read vectors -> output
    k_rv<<<Bq, 256, 0, stream>>>(out + O_WR, out + O_MEM, out + O_RV);

    // 14) final projection
    k_hv<<<(Bq * (Hq + Rq * Wq)) / 256, 256, 0, stream>>>(ws + HA, out + O_RV, ws + HV);
    dim3 gO(Wq / 16, Bq / 16);  // (4, 4)
    wmma_gemm<<<gO, 32, 0, stream>>>(ws + HV, Wout, bout, nullptr, out + O_OUT,
                                     Bq, Wq, Hq + Rq * Wq);
}
